// RNN_73761768342006
// MI455X (gfx1250) — matrix operand order repaired and checked by reading, not compiled, not run on hardware
//
#include <hip/hip_runtime.h>
#include <hip/hip_bf16.h>

typedef __attribute__((ext_vector_type(16))) _Float16 v16h;
typedef __attribute__((ext_vector_type(8)))  _Float16 v8h;
typedef __attribute__((ext_vector_type(8)))  float    v8f;
typedef __attribute__((ext_vector_type(4)))  float    v4f;

constexpr int Bb = 128, Ss = 2048, Dd = 128, Hh = 128;
constexpr int LDP = 136;               // padded LDS row stride in halves (272B, 16B aligned, bank-spread)
constexpr int RTOT = Bb * Ss;          // 262144 rows for the two big GEMMs
constexpr int NTILES = RTOT / 16;      // 16384 row tiles

// Workgroup barrier that only orders LDS traffic (avoids __syncthreads' full
// fence, which stalls on s_wait_storecnt for fire-and-forget global stores).
__device__ __forceinline__ void wg_barrier_lds() {
  asm volatile("s_wait_dscnt 0x0" ::: "memory");
  asm volatile("s_barrier_signal -1\n\ts_barrier_wait -1" ::: "memory");
}

// Barrier after async-to-LDS staging (ASYNCcnt tracks the DMA writes).
__device__ __forceinline__ void wg_barrier_async() {
  asm volatile("s_wait_asynccnt 0x0" ::: "memory");
  asm volatile("s_barrier_signal -1\n\ts_barrier_wait -1" ::: "memory");
}

// Build a 16x32-f16 A fragment from an LDS row (row-major, k0 = chunk base).
// Halves 0..7  <- local K = kg*16 + 0..7      (global k0..k0+7)
// Halves 8..15 <- local K = kg*16 + 8..15     (global k0+8..k0+15)
__device__ __forceinline__ v16h make_a(const _Float16* row, int k0) {
  v8h lo = *(const v8h*)(row + k0);
  v8h hi = *(const v8h*)(row + k0 + 8);
  return __builtin_shufflevector(lo, hi, 0,1,2,3,4,5,6,7,8,9,10,11,12,13,14,15);
}

// Branch-free tanh, exact at saturation: 1 - 2/(exp(2x)+1).
__device__ __forceinline__ float fast_tanh(float x) {
  float e = __expf(2.0f * x);
  return 1.0f - 2.0f / (e + 1.0f);
}

// ---------------------------------------------------------------------------
// Kernel 1: xp[r, n] = sum_k x[r,k] * W_ih[n,k] + (b_ih[n] + b_hh[n]), f16 out
// ---------------------------------------------------------------------------
__global__ void __launch_bounds__(256)
xp_gemm_f16(const float* __restrict__ x, const float* __restrict__ Wih,
            const float* __restrict__ b_ih, const float* __restrict__ b_hh,
            _Float16* __restrict__ xp)
{
  __shared__ __align__(16) _Float16 ax[16][LDP];
  const int tid  = threadIdx.x;
  const int lane = tid & 31;
  const int wave = tid >> 5;
  const int n    = lane & 15;       // C/D column, B column
  const int kg   = lane >> 4;       // K half selector
  const int n0   = wave * 16;       // this wave's N tile
  const int mA   = lane & 15;       // A row
  const int r    = tid >> 4;        // staging row
  const int ch   = tid & 15;        // staging chunk (8 elems)

  // Preload B fragments (W_ih rows) into registers, f32 -> f16.
  v16h bfrag[4];
  #pragma unroll
  for (int c4 = 0; c4 < 4; ++c4) {
    const float* wrow = Wih + (size_t)(n0 + n) * Dd + c4 * 32 + kg * 16;
    #pragma unroll
    for (int h = 0; h < 16; ++h) bfrag[c4][h] = (_Float16)wrow[h];
  }
  const float bias = b_ih[n0 + n] + b_hh[n0 + n];

  for (int tile = blockIdx.x; tile < NTILES; tile += gridDim.x) {
    const size_t row0 = (size_t)tile * 16;
    // Stage 16x128 f32 -> f16 into LDS (read-once x: non-temporal).
    const float* src = x + (row0 + r) * (size_t)Dd + ch * 8;
    v4f f0 = __builtin_nontemporal_load((const v4f*)src);
    v4f f1 = __builtin_nontemporal_load((const v4f*)(src + 4));
    v8h hx;
    #pragma unroll
    for (int i = 0; i < 4; ++i) { hx[i] = (_Float16)f0[i]; hx[4 + i] = (_Float16)f1[i]; }
    *(v8h*)&ax[r][ch * 8] = hx;
    wg_barrier_lds();

    // Gather all A fragments first so the 8 ds_loads pipeline.
    v16h a[4];
    #pragma unroll
    for (int c4 = 0; c4 < 4; ++c4) a[c4] = make_a(&ax[mA][0], c4 * 32 + kg * 16);

    v8f c;
    #pragma unroll
    for (int i = 0; i < 8; ++i) c[i] = bias;
    #pragma unroll
    for (int c4 = 0; c4 < 4; ++c4)
      c = __builtin_amdgcn_wmma_f32_16x16x32_f16(false, a[c4], false, bfrag[c4],
                                                 (short)0, c, false, false);
    #pragma unroll
    for (int i = 0; i < 8; ++i) {
      const int m = kg * 8 + i;
      xp[(row0 + m) * Hh + n0 + n] = (_Float16)c[i];
    }
    wg_barrier_lds();
  }
}

// ---------------------------------------------------------------------------
// Kernel 2: sequential scan. One workgroup per 16 batch rows; W_hh fragments
// in registers, h ping-pong in LDS (f16), per-lane xp values prefetched one
// full step ahead directly in C-fragment layout (L2-resident, no LDS trip).
// ---------------------------------------------------------------------------
__global__ void __launch_bounds__(256)
rnn_scan(const float* __restrict__ h0, const _Float16* __restrict__ xp,
         const float* __restrict__ Whh, _Float16* __restrict__ outh)
{
  __shared__ __align__(16) _Float16 hbuf[2][16][LDP];
  const int tid  = threadIdx.x;
  const int lane = tid & 31;
  const int wave = tid >> 5;
  const int n    = lane & 15;
  const int kg   = lane >> 4;
  const int n0   = wave * 16;
  const int mA   = lane & 15;
  const int r    = tid >> 4;
  const int ch   = tid & 15;
  const int b0   = blockIdx.x * 16;

  // W_hh B fragments: B[k][n] = W_hh[n][k]  (h_new = h @ W_hh^T)
  v16h bfrag[4];
  #pragma unroll
  for (int c4 = 0; c4 < 4; ++c4) {
    const float* wrow = Whh + (size_t)(n0 + n) * Hh + c4 * 32 + kg * 16;
    #pragma unroll
    for (int h = 0; h < 16; ++h) bfrag[c4][h] = (_Float16)wrow[h];
  }

  // Initial hidden state (f32 -> f16) into hbuf[0].
  {
    const float* src = h0 + (size_t)(b0 + r) * Hh + ch * 8;
    v8h hh;
    #pragma unroll
    for (int i = 0; i < 8; ++i) hh[i] = (_Float16)src[i];
    *(v8h*)&hbuf[0][r][ch * 8] = hh;
  }

  // Per-lane xp base in C-fragment layout: element i -> row m = kg*8+i, col n0+n.
  const size_t rowPitch = (size_t)Ss * Hh;                 // halves per batch row
  const _Float16* xpl = xp + ((size_t)(b0 + kg * 8) * Ss) * Hh + n0 + n;
  _Float16* outl = outh + ((size_t)(b0 + kg * 8) * Ss) * Hh + n0 + n;

  _Float16 cur[8];
  #pragma unroll
  for (int i = 0; i < 8; ++i) cur[i] = xpl[i * rowPitch];  // s = 0

  wg_barrier_lds();

  int p = 0;
  for (int s = 0; s < Ss; ++s) {
    // Prefetch next step's xp values (consumed next iteration -> full overlap).
    _Float16 nxt[8];
    if (s + 1 < Ss) {
      #pragma unroll
      for (int i = 0; i < 8; ++i) nxt[i] = xpl[i * rowPitch + (size_t)(s + 1) * Hh];
    }

    // Gather all A fragments (8 ds_load_b128 pipeline together).
    v16h a[4];
    #pragma unroll
    for (int c4 = 0; c4 < 4; ++c4) a[c4] = make_a(&hbuf[p][mA][0], c4 * 32 + kg * 16);

    // Two independent 2-deep WMMA chains halve the serial accumulate latency.
    v8f c0, c1;
    #pragma unroll
    for (int i = 0; i < 8; ++i) { c0[i] = (float)cur[i]; c1[i] = 0.0f; }
    c0 = __builtin_amdgcn_wmma_f32_16x16x32_f16(false, a[0], false, bfrag[0], (short)0, c0, false, false);
    c1 = __builtin_amdgcn_wmma_f32_16x16x32_f16(false, a[1], false, bfrag[1], (short)0, c1, false, false);
    c0 = __builtin_amdgcn_wmma_f32_16x16x32_f16(false, a[2], false, bfrag[2], (short)0, c0, false, false);
    c1 = __builtin_amdgcn_wmma_f32_16x16x32_f16(false, a[3], false, bfrag[3], (short)0, c1, false, false);

    // tanh + writeback (LDS ping-pong + global f16 history for the fc GEMM).
    #pragma unroll
    for (int i = 0; i < 8; ++i) {
      const int m = kg * 8 + i;
      const _Float16 hv = (_Float16)fast_tanh(c0[i] + c1[i]);
      hbuf[p ^ 1][m][n0 + n] = hv;
      outl[i * rowPitch + (size_t)s * Hh] = hv;
    }
    wg_barrier_lds();
    #pragma unroll
    for (int i = 0; i < 8; ++i) cur[i] = nxt[i];
    p ^= 1;
  }
}

// ---------------------------------------------------------------------------
// Kernel 3: out[r, o] = sum_h outh[r,h] * fc_w[o,h] + fc_b[o], f32 out (NT).
// Staging uses the gfx1250 async global->LDS DMA path (ASYNCcnt).
// ---------------------------------------------------------------------------
__global__ void __launch_bounds__(256)
fc_gemm(const _Float16* __restrict__ A, const float* __restrict__ fw,
        const float* __restrict__ fb, float* __restrict__ out)
{
  __shared__ __align__(16) _Float16 ax[16][LDP];
  const int tid  = threadIdx.x;
  const int lane = tid & 31;
  const int wave = tid >> 5;
  const int n    = lane & 15;
  const int kg   = lane >> 4;
  const int n0   = wave * 16;
  const int mA   = lane & 15;
  const int r    = tid >> 4;
  const int ch   = tid & 15;

  v16h bfrag[4];
  #pragma unroll
  for (int c4 = 0; c4 < 4; ++c4) {
    const float* wrow = fw + (size_t)(n0 + n) * Hh + c4 * 32 + kg * 16;
    #pragma unroll
    for (int h = 0; h < 16; ++h) bfrag[c4][h] = (_Float16)wrow[h];
  }
  const float bias = fb[n0 + n];

  // LDS byte offset of this thread's staging slot (ax is the only LDS object,
  // so it sits at segment offset 0).
  const unsigned ldsoff = (unsigned)(r * (LDP * 2) + ch * 16);

  for (int tile = blockIdx.x; tile < NTILES; tile += gridDim.x) {
    const size_t row0 = (size_t)tile * 16;
    // Async DMA 16B global -> LDS, no VGPR round trip; tracked by ASYNCcnt.
    const unsigned long long ga =
        (unsigned long long)(const void*)(A + (row0 + r) * (size_t)Hh + ch * 8);
    asm volatile("global_load_async_to_lds_b128 %0, %1, off"
                 :: "v"(ldsoff), "v"(ga) : "memory");
    wg_barrier_async();

    v16h a[4];
    #pragma unroll
    for (int c4 = 0; c4 < 4; ++c4) a[c4] = make_a(&ax[mA][0], c4 * 32 + kg * 16);

    v8f c;
    #pragma unroll
    for (int i = 0; i < 8; ++i) c[i] = bias;
    #pragma unroll
    for (int c4 = 0; c4 < 4; ++c4)
      c = __builtin_amdgcn_wmma_f32_16x16x32_f16(false, a[c4], false, bfrag[c4],
                                                 (short)0, c, false, false);
    #pragma unroll
    for (int i = 0; i < 8; ++i) {
      const int m = kg * 8 + i;
      __builtin_nontemporal_store(c[i], &out[(row0 + m) * Hh + n0 + n]);
    }
    wg_barrier_lds();
  }
}

extern "C" void kernel_launch(void* const* d_in, const int* in_sizes, int n_in,
                              void* d_out, int out_size, void* d_ws, size_t ws_size,
                              hipStream_t stream) {
  const float* hidden = (const float*)d_in[0];
  const float* x      = (const float*)d_in[1];
  const float* W_ih   = (const float*)d_in[2];
  const float* W_hh   = (const float*)d_in[3];
  const float* b_ih   = (const float*)d_in[4];
  const float* b_hh   = (const float*)d_in[5];
  const float* fc_w   = (const float*)d_in[6];
  const float* fc_b   = (const float*)d_in[7];
  float* out = (float*)d_out;

  // Workspace: xp (f16) + outh (f16), 64 MB each -> both stay L2-resident.
  _Float16* xp   = (_Float16*)d_ws;
  _Float16* outh = xp + (size_t)RTOT * Hh;

  xp_gemm_f16<<<2048, 256, 0, stream>>>(x, W_ih, b_ih, b_hh, xp);
  rnn_scan<<<Bb / 16, 256, 0, stream>>>(hidden, xp, W_hh, outh);
  fc_gemm<<<2048, 256, 0, stream>>>(outh, fc_w, fc_b, out);
}
